// STGNN_87479893885337
// MI455X (gfx1250) — matrix operand order repaired and checked by reading, not compiled, not run on hardware
//
#include <hip/hip_runtime.h>
#include <hip/hip_bf16.h>
#include <math.h>

// ---------------------------------------------------------------------------
// Problem constants (match reference)
// ---------------------------------------------------------------------------
#define Bq   32
#define TIN  12
#define TOUT 12
#define Nn   5000
#define Hh   64
#define Ee   80000

static constexpr size_t BN  = (size_t)Bq * Nn;    // 160000 rows (multiple of 16)
static constexpr size_t BNH = BN * Hh;            // 10,240,000 floats

// workspace layout (float offsets)
static constexpr size_t OFF_DIS  = 0;                 // N floats (deg then dis)
static constexpr size_t OFF_NORM = 16384;             // E floats
static constexpr size_t OFF_HLIN = 131072;            // B*N*H
static constexpr size_t OFF_H1   = OFF_HLIN + BNH;    // B*N*H (aliased by GRU output y)
static constexpr size_t OFF_H2   = OFF_H1 + BNH;      // B*N*H
static constexpr size_t OFF_GI   = OFF_H2 + BNH;      // B*N*192

typedef __attribute__((ext_vector_type(16))) _Float16 v16h;
typedef __attribute__((ext_vector_type(8)))  float    v8f;

// K-index mapping for 16-bit A fragments (16x32), per CDNA5 ISA 7.12.2:
// lanes 0-15 hold K={0..7,16..23}, lanes 16-31 hold K={8..15,24..31}
__device__ __forceinline__ int kmap16(int i, int g) {
    return ((i & 8) << 1) + (g << 3) + (i & 7);
}

__device__ __forceinline__ float sigmoidf_(float v) {
    return 1.0f / (1.0f + __expf(-v));
}

// ---------------------------------------------------------------------------
// Utility kernels
// ---------------------------------------------------------------------------
__global__ void zero_f32(float* __restrict__ p, size_t n) {
    size_t i = (size_t)blockIdx.x * blockDim.x + threadIdx.x;
    if (i < n) p[i] = 0.0f;
}

__global__ void deg_kernel(const int* __restrict__ dst, float* __restrict__ deg) {
    int e = blockIdx.x * blockDim.x + threadIdx.x;
    if (e < Ee) atomicAdd(&deg[dst[e]], 1.0f);
}

// dis[i] = (deg[i] + 1 [self loop])^-0.5   (always > 0)
__global__ void dis_kernel(float* __restrict__ deg_dis) {
    int i = blockIdx.x * blockDim.x + threadIdx.x;
    if (i < Nn) deg_dis[i] = rsqrtf(deg_dis[i] + 1.0f);
}

__global__ void norm_kernel(const int* __restrict__ src, const int* __restrict__ dst,
                            const float* __restrict__ dis, float* __restrict__ norm) {
    int e = blockIdx.x * blockDim.x + threadIdx.x;
    if (e < Ee) norm[e] = dis[src[e]] * dis[dst[e]];
}

// ---------------------------------------------------------------------------
// GCN layer 1 linear: h_lin[b,n,f] = sum_t x[b,t,n] * W1[t,f]   (K=12: VALU)
// ---------------------------------------------------------------------------
__global__ void gcn1_lin(const float* __restrict__ x, const float* __restrict__ W1,
                         float* __restrict__ hlin) {
    size_t idx = (size_t)blockIdx.x * blockDim.x + threadIdx.x;
    if (idx >= BNH) return;
    int f = (int)(idx & 63);
    size_t bn = idx >> 6;               // b*N + n
    int n = (int)(bn % Nn);
    int b = (int)(bn / Nn);
    const float* xb = x + (size_t)b * TIN * Nn + n;
    float s = 0.0f;
#pragma unroll
    for (int t = 0; t < TIN; ++t)
        s = fmaf(xb[(size_t)t * Nn], W1[t * Hh + f], s);
    hlin[idx] = s;
}

// ---------------------------------------------------------------------------
// Edge aggregation: agg[b,dst,:] += h_lin[b,src,:] * norm[e]
// grid = (E, B/8), block = 64; each block processes 8 batches for one edge,
// reusing the edge indices/weight.
// ---------------------------------------------------------------------------
__global__ __launch_bounds__(64)
void agg_edges(const int* __restrict__ src, const int* __restrict__ dst,
               const float* __restrict__ norm, const float* __restrict__ hlin,
               float* __restrict__ agg) {
    int e = blockIdx.x;
    int f = threadIdx.x;
    int s = src[e], d = dst[e];
    float w = norm[e];
#pragma unroll
    for (int bb = 0; bb < 8; ++bb) {
        int b = blockIdx.y * 8 + bb;
        float v = hlin[((size_t)b * Nn + s) * Hh + f] * w;
        atomicAdd(&agg[((size_t)b * Nn + d) * Hh + f], v);
    }
}

// out = relu(agg + dis[n]^2 * h_lin + bias)   (self-loop folded in, in-place OK)
__global__ void gcn_finalize(float* __restrict__ agg, const float* __restrict__ hlin,
                             const float* __restrict__ dis, const float* __restrict__ bias) {
    size_t idx = (size_t)blockIdx.x * blockDim.x + threadIdx.x;
    if (idx >= BNH) return;
    int f = (int)(idx & 63);
    int n = (int)((idx >> 6) % Nn);
    float di = dis[n];
    float v = agg[idx] + di * di * hlin[idx] + bias[f];
    agg[idx] = v > 0.0f ? v : 0.0f;
}

// ---------------------------------------------------------------------------
// Templated WMMA GEMM: C[M,NC] = A[M,K](f32) * B + bias
//   TRANSB==0 : B = W[K,NC] row-major
//   TRANSB==1 : B = W[NC,K] row-major used transposed (B[k,j] = W[j,k])
// Weight staged once per block into LDS as f16, transposed & zero-padded to
// NCP = ceil16(NC): B fragments are contiguous LDS reads, no guards.
// Each wave owns one 16-row tile and CG (<=4) column tiles, reusing the A
// fragment across CG WMMAs. f32 accumulate via v_wmma_f32_16x16x32_f16.
// Requires: M % 16 == 0, K % 32 == 0.
// ---------------------------------------------------------------------------
template <int NC, int K, bool TRANSB, bool HASBIAS>
__global__ __launch_bounds__(128)
void wmma_gemm(const float* __restrict__ A, const float* __restrict__ W,
               const float* __restrict__ bias, float* __restrict__ C, int M) {
    constexpr int NCP      = (NC + 15) & ~15;        // padded columns
    constexpr int COLTILES = NCP / 16;
    constexpr int CG       = COLTILES >= 4 ? 4 : COLTILES;   // col tiles per wave
    constexpr int G        = COLTILES / CG;                  // col groups
    static_assert(COLTILES % CG == 0, "col tiling");
    static_assert(K % 32 == 0, "K tiling");

    __shared__ _Float16 sW[K * NCP];                 // <= 64*192*2 = 24 KB

    const int tid  = threadIdx.x;
    const int lane = tid & 31;
    const int warp = tid >> 5;

    // stage weights (transposed/padded) into LDS as f16
    for (int idx = tid; idx < K * NCP; idx += 128) {
        int k = idx / NCP;
        int c = idx - k * NCP;
        float wv = 0.0f;
        if (NC == NCP || c < NC)
            wv = TRANSB ? W[(size_t)c * K + k] : W[(size_t)k * NC + c];
        sW[idx] = (_Float16)wv;
    }
    __syncthreads();

    const int rowTiles = M >> 4;
    const int unit = blockIdx.x * 4 + warp;
    if (unit >= rowTiles * G) return;                // wave-uniform exit
    const int rt = unit / G;
    const int cg = unit - rt * G;
    const int cbase0 = cg * CG * 16;

    const int g = lane >> 4;                         // lane-half for A K-mapping
    const int r = lane & 15;                         // A row / C column within tile
    const int arow = rt * 16 + r;

    v8f acc[CG];
#pragma unroll
    for (int c = 0; c < CG; ++c) acc[c] = {};

    for (int kk = 0; kk < K; kk += 32) {
        // A fragment: two contiguous 8-float runs -> four b128 loads, cvt f16
        const float* ap = A + (size_t)arow * K + kk + (g << 3);
        float4 x0 = *(const float4*)(ap);
        float4 x1 = *(const float4*)(ap + 4);
        float4 x2 = *(const float4*)(ap + 16);
        float4 x3 = *(const float4*)(ap + 20);
        v16h a;
        a[0]  = (_Float16)x0.x; a[1]  = (_Float16)x0.y; a[2]  = (_Float16)x0.z; a[3]  = (_Float16)x0.w;
        a[4]  = (_Float16)x1.x; a[5]  = (_Float16)x1.y; a[6]  = (_Float16)x1.z; a[7]  = (_Float16)x1.w;
        a[8]  = (_Float16)x2.x; a[9]  = (_Float16)x2.y; a[10] = (_Float16)x2.z; a[11] = (_Float16)x2.w;
        a[12] = (_Float16)x3.x; a[13] = (_Float16)x3.y; a[14] = (_Float16)x3.z; a[15] = (_Float16)x3.w;

        const int kb = kk;                           // B fragment: lane = N column, K via kmap16
#pragma unroll
        for (int c = 0; c < CG; ++c) {
            const _Float16* bp = &sW[kb * NCP + cbase0 + c * 16 + r];
            v16h bf;
#pragma unroll
            for (int i = 0; i < 16; ++i) bf[i] = bp[kmap16(i, g) * NCP];
            acc[c] = __builtin_amdgcn_wmma_f32_16x16x32_f16(
                false, a, false, bf, (short)0, acc[c], false, false);
        }
    }

    // C/D layout: lane = N column, VGPR v = row (v + 8*g)
#pragma unroll
    for (int c = 0; c < CG; ++c) {
        int col = cbase0 + c * 16 + r;
        if (NC != NCP && col >= NC) continue;        // folds away for NC%16==0
        float bv = HASBIAS ? bias[col] : 0.0f;
#pragma unroll
        for (int v = 0; v < 8; ++v) {
            int row = rt * 16 + v + (g << 3);
            C[(size_t)row * NC + col] = acc[c][v] + bv;
        }
    }
}

// ---------------------------------------------------------------------------
// GRU sequential scan over node dim (length N), state [B=32, H=64].
// gi[b,t,:] precomputed (= x_t @ w_ih.T + b_ih).  Per step:
//   issue this step's gi loads into registers (latency hidden behind WMMA),
//   gh = h_prev @ w_hh.T + b_hh   via 24 wmma tiles (2 Mtiles x 12 Ntiles),
//   gates r,z,n; h_new; write y[b,t,:].
// One workgroup, 8 waves (3 tiles each); w_hh.T (f16), h (f32+f16 mirror)
// and gh live in LDS. ~60 KB LDS. gi (123 MB) fits the 192 MB L2, so a
// global_prefetch 16 steps ahead keeps the scan at L2 latency.
// ---------------------------------------------------------------------------
__global__ __launch_bounds__(256)
void gru_scan(const float* __restrict__ gi, const float* __restrict__ w_hh,
              const float* __restrict__ b_hh, float* __restrict__ y) {
    __shared__ _Float16 sB[64 * 192];     // B[k,j] = w_hh[j,k], f16
    __shared__ float    sH[32 * 64];      // h_prev f32
    __shared__ _Float16 sH16[32 * 64];    // h_prev f16 mirror (WMMA A)
    __shared__ float    sGH[32 * 192];    // gh staging

    const int tid  = threadIdx.x;
    const int lane = tid & 31;
    const int warp = tid >> 5;
    const int g    = lane >> 4;
    const int r    = lane & 15;

    // stage w_hh.T into LDS as f16 (coalesced read of w_hh[192,64])
    for (int idx = tid; idx < 192 * 64; idx += 256) {
        int j = idx >> 6;                 // row of w_hh  -> column of B
        int k = idx & 63;                 // col of w_hh  -> row of B
        sB[k * 192 + j] = (_Float16)w_hh[idx];
    }
    for (int idx = tid; idx < 32 * 64; idx += 256) {
        sH[idx] = 0.0f;
        sH16[idx] = (_Float16)0.0f;
    }
    __syncthreads();

    for (int t = 0; t < Nn; ++t) {
        // --- issue gi loads for this step early (software pipelining) -----
        float gr[8], gz[8], gn[8];
#pragma unroll
        for (int it = 0; it < 8; ++it) {
            int u = tid + (it << 8);
            const float* gp = gi + ((size_t)(u >> 6) * Nn + t) * 192 + (u & 63);
            gr[it] = gp[0];
            gz[it] = gp[64];
            gn[it] = gp[128];
        }
        if (t + 16 < Nn) {                // warm L2 for a future step
            __builtin_prefetch(gi + ((size_t)(tid >> 6) * Nn + (t + 16)) * 192 + (tid & 63), 0, 1);
        }

        // --- gh = h_prev @ w_hh.T + b_hh ----------------------------------
#pragma unroll
        for (int it = 0; it < 3; ++it) {
            int p  = warp + (it << 3);    // tile 0..23
            int mt = p / 12;
            int nt = p - mt * 12;
            int arow = mt * 16 + r;
            v8f acc = {};
#pragma unroll
            for (int kk = 0; kk < 64; kk += 32) {
                v16h a, bf;
#pragma unroll
                for (int i = 0; i < 16; ++i)
                    a[i] = sH16[arow * 64 + kk + kmap16(i, g)];
                int kb = kk;
#pragma unroll
                for (int i = 0; i < 16; ++i)
                    bf[i] = sB[(kb + kmap16(i, g)) * 192 + nt * 16 + r];
                acc = __builtin_amdgcn_wmma_f32_16x16x32_f16(
                    false, a, false, bf, (short)0, acc, false, false);
            }
            int col = nt * 16 + r;
            float bv = b_hh[col];
#pragma unroll
            for (int v = 0; v < 8; ++v) {
                int row = mt * 16 + v + (g << 3);
                sGH[row * 192 + col] = acc[v] + bv;
            }
        }
        __syncthreads();

        // --- gates + state update (2048 elements over 256 threads) --------
#pragma unroll
        for (int it = 0; it < 8; ++it) {
            int u = tid + (it << 8);
            int b = u >> 6;
            int j = u & 63;
            float hr = sGH[b * 192 + j];
            float hz = sGH[b * 192 + 64 + j];
            float hn = sGH[b * 192 + 128 + j];
            float rg = sigmoidf_(gr[it] + hr);
            float zg = sigmoidf_(gz[it] + hz);
            float ng = tanhf(gn[it] + rg * hn);
            float hp = sH[b * 64 + j];
            float hnew = (1.0f - zg) * ng + zg * hp;
            sH[b * 64 + j]   = hnew;
            sH16[b * 64 + j] = (_Float16)hnew;
            y[((size_t)b * Nn + t) * 64 + j] = hnew;
        }
        __syncthreads();   // sH16 ready / sGH consumed before next step
    }
}

// ---------------------------------------------------------------------------
// Host-side launch sequence
// ---------------------------------------------------------------------------
extern "C" void kernel_launch(void* const* d_in, const int* in_sizes, int n_in,
                              void* d_out, int out_size, void* d_ws, size_t ws_size,
                              hipStream_t stream) {
    const float* x    = (const float*)d_in[0];
    const int*   ei   = (const int*)d_in[1];
    const float* W1   = (const float*)d_in[2];
    const float* b1   = (const float*)d_in[3];
    const float* W2   = (const float*)d_in[4];
    const float* b2   = (const float*)d_in[5];
    const float* w_ih = (const float*)d_in[6];
    const float* w_hh = (const float*)d_in[7];
    const float* b_ih = (const float*)d_in[8];
    const float* b_hh = (const float*)d_in[9];
    const float* Wfc  = (const float*)d_in[10];
    const float* bfc  = (const float*)d_in[11];
    (void)in_sizes; (void)n_in; (void)out_size; (void)ws_size;

    const int* src = ei;
    const int* dst = ei + Ee;

    float* ws   = (float*)d_ws;
    float* dis  = ws + OFF_DIS;
    float* norm = ws + OFF_NORM;
    float* hlin = ws + OFF_HLIN;
    float* h1   = ws + OFF_H1;
    float* h2   = ws + OFF_H2;
    float* gi   = ws + OFF_GI;
    float* y    = h1;                    // h1 dead once h2 is final

    const int THR = 256;
    auto blocks = [](size_t n, int t) { return (unsigned)((n + t - 1) / t); };

    // --- GCN normalization -------------------------------------------------
    zero_f32<<<blocks(Nn, THR), THR, 0, stream>>>(dis, Nn);
    deg_kernel<<<blocks(Ee, THR), THR, 0, stream>>>(dst, dis);
    dis_kernel<<<blocks(Nn, THR), THR, 0, stream>>>(dis);
    norm_kernel<<<blocks(Ee, THR), THR, 0, stream>>>(src, dst, dis, norm);

    // --- GCN layer 1 -------------------------------------------------------
    gcn1_lin<<<blocks(BNH, THR), THR, 0, stream>>>(x, W1, hlin);
    zero_f32<<<blocks(BNH, THR), THR, 0, stream>>>(h1, BNH);
    agg_edges<<<dim3(Ee, Bq / 8), 64, 0, stream>>>(src, dst, norm, hlin, h1);
    gcn_finalize<<<blocks(BNH, THR), THR, 0, stream>>>(h1, hlin, dis, b1);

    // --- GCN layer 2 linear (WMMA): units = rowTiles * 1 -------------------
    wmma_gemm<Hh, Hh, false, false>
        <<<blocks(BN / 16, 4), 128, 0, stream>>>(h1, W2, nullptr, hlin, (int)BN);
    zero_f32<<<blocks(BNH, THR), THR, 0, stream>>>(h2, BNH);
    agg_edges<<<dim3(Ee, Bq / 8), 64, 0, stream>>>(src, dst, norm, hlin, h2);
    gcn_finalize<<<blocks(BNH, THR), THR, 0, stream>>>(h2, hlin, dis, b2);

    // --- GRU input gates hoisted out of the scan (WMMA): units = rowTiles*3
    wmma_gemm<3 * Hh, Hh, true, true>
        <<<blocks((BN / 16) * 3, 4), 128, 0, stream>>>(h2, w_ih, b_ih, gi, (int)BN);

    // --- sequential GRU scan (single WGP, WMMA recurrent GEMM) -------------
    gru_scan<<<1, 256, 0, stream>>>(gi, w_hh, b_hh, y);

    // --- final FC (WMMA, Nc=12 via zero-padded LDS tile) --------------------
    wmma_gemm<TOUT, Hh, true, true>
        <<<blocks(BN / 16, 4), 128, 0, stream>>>(y, Wfc, bfc, (float*)d_out, (int)BN);
}
